// LocalAttention_41068477285011
// MI455X (gfx1250) — matrix operand order repaired and checked by reading, not compiled, not run on hardware
//
#include <hip/hip_runtime.h>

// ---------------------------------------------------------------------------
// Local (sliding-window, causal) attention block for MI455X / gfx1250.
// B=2, T=2048, D=1024, H=16, HD=64, W=256.
//   - fp32 inputs converted to bf16 ONCE (elementwise pass).
//   - All GEMM-like math on v_wmma_f32_16x16x32_bf16 (fp32 accum).
//   - GEMM tiles staged cache->LDS with global_load_async_to_lds_b128
//     (ASYNCcnt path), fragments read via ds_load.
//   - Intermediates (qkv, y) kept in bf16: halves streaming bytes.
//   - Attention K/V tail rows handled by ADDRESS CLAMPING (mask kills those
//     columns), so fragment loads are branchless straight-line code.
// Workspace (bf16): x 8MB + Wqkv 6MB + Wo 2MB + qkv 24MB + y 8MB = 48MB.
// ---------------------------------------------------------------------------

#define T_SEQ   2048
#define D_MODEL 1024
#define NHEAD   16
#define WSIZE   256
#define HDIM    64
#define QKV_STRIDE (3 * D_MODEL)

typedef __attribute__((ext_vector_type(16))) __bf16 v16bf;
typedef __attribute__((ext_vector_type(8)))  float  v8f;

union FragAB { v16bf v; unsigned u[8]; };
union FragC  { v8f   v; float    f[8]; };

// fp32 -> bf16, round-to-nearest-even
__device__ __forceinline__ unsigned short f2bf(float f) {
  unsigned u = __builtin_bit_cast(unsigned, f);
  u += 0x7FFFu + ((u >> 16) & 1u);
  return (unsigned short)(u >> 16);
}
__device__ __forceinline__ unsigned pack2(float lo, float hi) {
  return (unsigned)f2bf(lo) | ((unsigned)f2bf(hi) << 16);
}

// ISA 16-bit A-matrix 16x32 layout: lanes 0-15 -> K 0..7 / 16..23,
// lanes 16-31 -> K 8..15 / 24..31, two bf16 per VGPR.
__device__ __forceinline__ int a_k0(int v, int h) {
  return (v < 4) ? (h * 8 + v * 2) : (16 + h * 8 + (v - 4) * 2);
}
// 16-bit B-matrix 32x16 layout (same k order as A): lanes 0-15 -> K 0..7 / 16..23,
// lanes 16-31 -> K 8..15 / 24..31.
__device__ __forceinline__ int b_k0(int v, int h) { return a_k0(v, h); }

__device__ __forceinline__ v8f wmma_bf16(const FragAB& a, const FragAB& b, v8f c) {
  return __builtin_amdgcn_wmma_f32_16x16x32_bf16(
      false, a.v, false, b.v, (short)0, c, false, false);
}

// Async cache->LDS copy, 16 bytes per lane.  VDST = wave-relative LDS byte
// address (low 32 bits of the flat shared pointer), ADDR = 64-bit global.
__device__ __forceinline__ void async_b128(unsigned lds_off, const void* gptr) {
  asm volatile("global_load_async_to_lds_b128 %0, %1, off"
               :: "v"(lds_off), "v"((unsigned long long)(uintptr_t)gptr)
               : "memory");
}
__device__ __forceinline__ void wait_async0() {
  asm volatile("s_wait_asynccnt 0x0" ::: "memory");
}

// ---------------------------------------------------------------------------
// One-time fp32 -> bf16 conversion, 4 elements/thread.
// ---------------------------------------------------------------------------
__global__ __launch_bounds__(256) void cvt_bf16(
    const float* __restrict__ in, unsigned short* __restrict__ out, int n4)
{
  int i = blockIdx.x * 256 + threadIdx.x;
  if (i < n4) {
    float4 f = ((const float4*)in)[i];
    uint2 p;
    p.x = pack2(f.x, f.y);
    p.y = pack2(f.z, f.w);
    ((uint2*)out)[i] = p;
  }
}

// ---------------------------------------------------------------------------
// C[M,N] = A[M,K] * Wt[N,K]^T + bias[N]   (bf16 in, bf16 or fp32 out)
// 256 threads = 8 waves (2x4), block tile 64x128, K-step 32.
// Tiles staged with global_load_async_to_lds_b128; 80B LDS rows are both
// 16B-aligned for async writes and bank-conflict-free for fragment reads.
// ---------------------------------------------------------------------------
template <bool OUT_BF16>
__global__ __launch_bounds__(256) void gemm_bias_wmma(
    const unsigned short* __restrict__ A, const unsigned short* __restrict__ Wt,
    const float* __restrict__ bias, void* __restrict__ Cout,
    int M, int N, int K)
{
  constexpr int BM = 64, BN = 128, BK = 32, LDK = 40;  // 40 ushorts = 80B rows
  __shared__ unsigned short As[BM][LDK];
  __shared__ unsigned short Bs[BN][LDK];

  const int tid  = threadIdx.x;
  const int bm   = blockIdx.y * BM;
  const int bn   = blockIdx.x * BN;
  const int wave = tid >> 5, lane = tid & 31;
  const int wm   = wave >> 2, wn = wave & 3;
  const int n16  = lane & 15, hh = lane >> 4;

  FragC acc[2][2];
  #pragma unroll
  for (int mi = 0; mi < 2; ++mi)
    #pragma unroll
    for (int ni = 0; ni < 2; ++ni)
      #pragma unroll
      for (int v = 0; v < 8; ++v) acc[mi][ni].f[v] = 0.f;

  for (int kk = 0; kk < K; kk += BK) {
    // ---- async stage A (64x32 bf16): 1 x b128 per thread ----
    {
      int r = tid >> 2, s = tid & 3;  // 4 x 16B segments per 64B row
      async_b128((unsigned)(uintptr_t)&As[r][s * 8],
                 A + (size_t)(bm + r) * K + kk + s * 8);
    }
    // ---- async stage B (128x32 bf16): 2 x b128 per thread ----
    #pragma unroll
    for (int i = 0; i < 2; ++i) {
      int slot = tid + i * 256;
      int r = slot >> 2, s = slot & 3;
      async_b128((unsigned)(uintptr_t)&Bs[r][s * 8],
                 Wt + (size_t)(bn + r) * K + kk + s * 8);
    }
    if (kk + BK < K) {
      __builtin_prefetch(A + (size_t)(bm + (tid >> 2)) * K + kk + BK, 0, 1);
      __builtin_prefetch(Wt + (size_t)(bn + (tid >> 1)) * K + kk + BK, 0, 1);
    }
    wait_async0();
    __syncthreads();

    FragAB a[2], b[2];
    #pragma unroll
    for (int mi = 0; mi < 2; ++mi) {
      int r = wm * 32 + mi * 16 + n16;
      #pragma unroll
      for (int v = 0; v < 8; ++v) a[mi].u[v] = *(const unsigned*)&As[r][a_k0(v, hh)];
    }
    #pragma unroll
    for (int ni = 0; ni < 2; ++ni) {
      int r = wn * 32 + ni * 16 + n16;
      #pragma unroll
      for (int v = 0; v < 8; ++v) b[ni].u[v] = *(const unsigned*)&Bs[r][b_k0(v, hh)];
    }
    #pragma unroll
    for (int mi = 0; mi < 2; ++mi)
      #pragma unroll
      for (int ni = 0; ni < 2; ++ni)
        acc[mi][ni].v = wmma_bf16(a[mi], b[ni], acc[mi][ni].v);
    __syncthreads();
  }

  // C-fragment layout: row = v + 8*half, col = lane&15.
  #pragma unroll
  for (int mi = 0; mi < 2; ++mi)
    #pragma unroll
    for (int ni = 0; ni < 2; ++ni)
      #pragma unroll
      for (int v = 0; v < 8; ++v) {
        int r = bm + wm * 32 + mi * 16 + v + 8 * hh;
        int c = bn + wn * 32 + ni * 16 + n16;
        float val = acc[mi][ni].f[v] + bias[c];
        if constexpr (OUT_BF16)
          ((unsigned short*)Cout)[(size_t)r * N + c] = f2bf(val);
        else
          ((float*)Cout)[(size_t)r * N + c] = val;
      }
}

// ---------------------------------------------------------------------------
// Sliding-window causal attention, flash-style online softmax, bf16 I/O.
// One wave per 16-row query tile; kv consumed in 32-wide chunks.
// qkv layout: [b, t, c*D + head*HD + hd] bf16.  y: [b, t, head*HD+hd] bf16.
// Out-of-range kv rows are CLAMPED, not predicated: the sliding-window mask
// (for K) and the exactly-zero masked probabilities (for V) make the loaded
// values irrelevant, so all fragment loads are branchless.
// ---------------------------------------------------------------------------
__global__ __launch_bounds__(256) void attn_sliding_wmma(
    const unsigned short* __restrict__ qkv, unsigned short* __restrict__ y)
{
  __shared__ unsigned short Plds[8][16][34];  // per-wave P staging (C -> A layout)

  const int tid   = threadIdx.x;
  const int wslot = tid >> 5;
  const int lane  = tid & 31;
  const int n16   = lane & 15, hh = lane >> 4;
  const int wid   = blockIdx.x * 8 + wslot;   // 0 .. B*H*(T/16)-1
  const int qt    = wid & 127;                // T/16 = 128
  const int bh    = wid >> 7;
  const int b     = bh >> 4, head = bh & 15;
  const int qs    = qt * 16;

  const unsigned short* base = qkv + (size_t)b * T_SEQ * QKV_STRIDE + head * HDIM;
  const unsigned short* qb = base;
  const unsigned short* kb = base + D_MODEL;
  const unsigned short* vb = base + 2 * D_MODEL;

  // Q fragments: 16x64 as two 16x32 A-fragments (pairs are contiguous bf16).
  FragAB qa[2];
  #pragma unroll
  for (int c = 0; c < 2; ++c)
    #pragma unroll
    for (int v = 0; v < 8; ++v) {
      int hd = c * 32 + a_k0(v, hh);
      qa[c].u[v] = *(const unsigned*)(qb + (size_t)(qs + n16) * QKV_STRIDE + hd);
    }

  FragC acc[4];
  float m_run[8], l_run[8];
  #pragma unroll
  for (int t = 0; t < 4; ++t)
    #pragma unroll
    for (int v = 0; v < 8; ++v) acc[t].f[v] = 0.f;
  #pragma unroll
  for (int v = 0; v < 8; ++v) { m_run[v] = -1e30f; l_run[v] = 0.f; }

  const float scale = 0.125f;  // 1/sqrt(HD)
  int kc0 = qs - (WSIZE - 1);
  if (kc0 < 0) kc0 = 0;
  kc0 &= ~31;

  for (int kc = kc0; kc <= qs + 15; kc += 32) {
    // ---- S = Q * K^T for this 16x32 score slab (two 16x16 C tiles) ----
    FragC s[2];
    #pragma unroll
    for (int st = 0; st < 2; ++st) {
      #pragma unroll
      for (int v = 0; v < 8; ++v) s[st].f[v] = 0.f;
      FragAB kf[2];
      int j  = kc + st * 16 + n16;                    // kv row (B-matrix column)
      int jc = (j < T_SEQ - 1) ? j : (T_SEQ - 1);     // clamp; mask kills j>=T
      const unsigned short* krow = kb + (size_t)jc * QKV_STRIDE;
      #pragma unroll
      for (int c = 0; c < 2; ++c)
        #pragma unroll
        for (int v = 0; v < 8; ++v)
          kf[c].u[v] = *(const unsigned*)(krow + c * 32 + b_k0(v, hh));
      s[st].v = wmma_bf16(qa[0], kf[0], s[st].v);
      s[st].v = wmma_bf16(qa[1], kf[1], s[st].v);
    }

    // ---- mask + scale + online softmax (rows live on vgpr v + 8*half) ----
    float rm[8];
    #pragma unroll
    for (int v = 0; v < 8; ++v) {
      int i_g = qs + v + 8 * hh;
      float best = -1e30f;
      #pragma unroll
      for (int st = 0; st < 2; ++st) {
        int j_g = kc + st * 16 + n16;
        float sv = s[st].f[v] * scale;
        bool keep = (j_g <= i_g) && (j_g >= i_g - (WSIZE - 1));
        sv = keep ? sv : -1e30f;
        s[st].f[v] = sv;
        best = fmaxf(best, sv);
      }
      rm[v] = best;
    }
    #pragma unroll
    for (int off = 8; off > 0; off >>= 1)
      #pragma unroll
      for (int v = 0; v < 8; ++v)
        rm[v] = fmaxf(rm[v], __shfl_xor(rm[v], off, 16));

    float corr[8], rs[8];
    #pragma unroll
    for (int v = 0; v < 8; ++v) {
      float mn = fmaxf(m_run[v], rm[v]);
      corr[v] = __expf(m_run[v] - mn);
      m_run[v] = mn;
      rs[v] = 0.f;
    }
    #pragma unroll
    for (int st = 0; st < 2; ++st)
      #pragma unroll
      for (int v = 0; v < 8; ++v) {
        float p = __expf(s[st].f[v] - m_run[v]);   // masked -> exactly 0
        s[st].f[v] = p;
        rs[v] += p;
      }
    #pragma unroll
    for (int off = 8; off > 0; off >>= 1)
      #pragma unroll
      for (int v = 0; v < 8; ++v)
        rs[v] += __shfl_xor(rs[v], off, 16);
    #pragma unroll
    for (int v = 0; v < 8; ++v) l_run[v] = l_run[v] * corr[v] + rs[v];
    #pragma unroll
    for (int t = 0; t < 4; ++t)
      #pragma unroll
      for (int v = 0; v < 8; ++v) acc[t].f[v] *= corr[v];

    // ---- P: C-layout -> A-layout via wave-private LDS patch ----
    #pragma unroll
    for (int st = 0; st < 2; ++st)
      #pragma unroll
      for (int v = 0; v < 8; ++v)
        Plds[wslot][v + 8 * hh][st * 16 + n16] = f2bf(s[st].f[v]);
    asm volatile("s_wait_dscnt 0" ::: "memory");  // wave-private LDS RAW
    FragAB pa;
    #pragma unroll
    for (int v = 0; v < 8; ++v)
      pa.u[v] = *(const unsigned*)&Plds[wslot][n16][a_k0(v, hh)];

    // ---- acc += P * V (four 16x16 hd tiles); OOB rows have P == 0 ----
    #pragma unroll
    for (int ht = 0; ht < 4; ++ht) {
      FragAB vf;
      int hd = ht * 16 + n16;
      #pragma unroll
      for (int v = 0; v < 8; ++v) {
        int j0 = kc + b_k0(v, hh);
        int j0c = (j0     < T_SEQ - 1) ? j0       : (T_SEQ - 1);
        int j1c = (j0 + 1 < T_SEQ - 1) ? (j0 + 1) : (T_SEQ - 1);
        unsigned lo = vb[(size_t)j0c * QKV_STRIDE + hd];
        unsigned hi = vb[(size_t)j1c * QKV_STRIDE + hd];
        vf.u[v] = lo | (hi << 16);
      }
      acc[ht].v = wmma_bf16(pa, vf, acc[ht].v);
    }
  }

  // ---- normalize + store y (bf16) ----
  #pragma unroll
  for (int ht = 0; ht < 4; ++ht)
    #pragma unroll
    for (int v = 0; v < 8; ++v) {
      int t  = qs + v + 8 * hh;
      int hd = ht * 16 + n16;
      y[((size_t)b * T_SEQ + t) * D_MODEL + head * HDIM + hd] =
          f2bf(acc[ht].f[v] / l_run[v]);
    }
}

// ---------------------------------------------------------------------------
extern "C" void kernel_launch(void* const* d_in, const int* in_sizes, int n_in,
                              void* d_out, int out_size, void* d_ws, size_t ws_size,
                              hipStream_t stream) {
  const float* x    = (const float*)d_in[0];
  const float* Wqkv = (const float*)d_in[1];
  const float* bqkv = (const float*)d_in[2];
  const float* Wo   = (const float*)d_in[3];
  const float* bo   = (const float*)d_in[4];
  float* out = (float*)d_out;

  const int M = 2 * T_SEQ;  // B*T = 4096 rows

  unsigned short* xbf    = (unsigned short*)d_ws;                 // 4096x1024
  unsigned short* wqkvbf = xbf    + (size_t)M * D_MODEL;          // 3072x1024
  unsigned short* wobf   = wqkvbf + (size_t)QKV_STRIDE * D_MODEL; // 1024x1024
  unsigned short* qkvbf  = wobf   + (size_t)D_MODEL * D_MODEL;    // 4096x3072
  unsigned short* ybf    = qkvbf  + (size_t)M * QKV_STRIDE;       // 4096x1024

  // 0) one-time fp32 -> bf16 conversions
  cvt_bf16<<<dim3((M * D_MODEL / 4) / 256), 256, 0, stream>>>(x, xbf, M * D_MODEL / 4);
  cvt_bf16<<<dim3((QKV_STRIDE * D_MODEL / 4) / 256), 256, 0, stream>>>(
      Wqkv, wqkvbf, QKV_STRIDE * D_MODEL / 4);
  cvt_bf16<<<dim3((D_MODEL * D_MODEL / 4) / 256), 256, 0, stream>>>(
      Wo, wobf, D_MODEL * D_MODEL / 4);

  // 1) qkv = x @ Wqkv^T + bqkv   (bf16 out)
  gemm_bias_wmma<true><<<dim3(QKV_STRIDE / 128, M / 64), 256, 0, stream>>>(
      xbf, wqkvbf, bqkv, qkvbf, M, QKV_STRIDE, D_MODEL);

  // 2) windowed causal attention -> y (bf16)
  attn_sliding_wmma<<<dim3((2 * NHEAD * (T_SEQ / 16)) / 8), 256, 0, stream>>>(
      qkvbf, ybf);

  // 3) out = y @ Wo^T + bo   (fp32 out)
  gemm_bias_wmma<false><<<dim3(D_MODEL / 128, M / 64), 256, 0, stream>>>(
      ybf, wobf, bo, out, M, D_MODEL, D_MODEL);
}
